// MLP_Adapter_1726576853558
// MI455X (gfx1250) — matrix operand order repaired and checked by reading, not compiled, not run on hardware
//
#include <hip/hip_runtime.h>
#include <math.h>

// ---------------------------------------------------------------------------
// Types for CDNA5 WMMA (wave32): 16x16x32 bf16 -> f32
// ---------------------------------------------------------------------------
typedef __attribute__((ext_vector_type(16))) __bf16 v16bf;
typedef __attribute__((ext_vector_type(8)))  float  v8f;

union V16A { v16bf v; unsigned int u[8]; };

__device__ __forceinline__ unsigned short f32_to_bf16(float f) {
  unsigned int x = __float_as_uint(f);
  unsigned int r = (x + 0x7FFFu + ((x >> 16) & 1u)) >> 16;  // RNE
  return (unsigned short)r;
}
__device__ __forceinline__ float bf16_to_f32(unsigned short u) {
  return __uint_as_float(((unsigned int)u) << 16);
}

// ---------------------------------------------------------------------------
// fp32 -> bf16 cast (grid-stride)
// ---------------------------------------------------------------------------
__global__ void cast_f32_bf16(const float* __restrict__ src,
                              unsigned short* __restrict__ dst, int n) {
  int i = blockIdx.x * blockDim.x + threadIdx.x;
  int stride = gridDim.x * blockDim.x;
  for (; i < n; i += stride) dst[i] = f32_to_bf16(src[i]);
}

// ---------------------------------------------------------------------------
// WMMA GEMM: C[M,N] = A[M,K] @ Bw[N,K]^T (+ bias) with fused epilogues.
// One wave per 32x32 C tile (2x2 register blocking -> 4 WMMAs per K-step,
// 2x the arithmetic intensity of a 16x16 tile).  K-loop in steps of 32 using
// v_wmma_f32_16x16x32_bf16.  Lane layouts per CDNA5 ISA 7.12.2.
// MODE 0: out f32 + out bf16                  (fc1 -> xs)
// MODE 1: out bf16                            (convA / qkv / proj)
// MODE 2: +residual f32, exact GELU, out bf16 (convB)
// MODE 5: out f32 = addF + acc + bias         (fc2 + final residual -> d_out)
// ---------------------------------------------------------------------------
template <int MODE>
__global__ void wmma_gemm(const unsigned short* __restrict__ A,
                          const unsigned short* __restrict__ Bw,
                          const float* __restrict__ bias,
                          int M, int N, int K,
                          float* __restrict__ outF,
                          unsigned short* __restrict__ outB,
                          const float* __restrict__ addF) {
  const int wave = threadIdx.x >> 5;
  const int lane = threadIdx.x & 31;
  const int l16  = lane & 15;
  const int hb   = lane >> 4;
  const int ntiles = N >> 5;                  // 32-wide C tiles
  const int tile = blockIdx.x * (blockDim.x >> 5) + wave;
  if (tile >= (M >> 5) * ntiles) return;      // wave-uniform: EXEC stays all-1s
  const int tm = tile / ntiles;
  const int tn = tile - tm * ntiles;

  // A fragments: lane owns rows m = tm*32 + l16 (+16)
  // B fragments: lane owns weight rows n = tn*32 + l16 (+16)
  const unsigned short* arow0 = A  + (size_t)(tm * 32 + l16) * K;
  const unsigned short* arow1 = arow0 + (size_t)16 * K;
  const unsigned short* brow0 = Bw + (size_t)(tn * 32 + l16) * K + hb * 16;
  const unsigned short* brow1 = brow0 + (size_t)16 * K;

  v8f acc[2][2];
#pragma unroll
  for (int i = 0; i < 2; ++i)
#pragma unroll
    for (int j = 0; j < 2; ++j)
      acc[i][j] = v8f{0.f, 0.f, 0.f, 0.f, 0.f, 0.f, 0.f, 0.f};

  for (int k0 = 0; k0 < K; k0 += 32) {
    // A 16x32 bf16 layout: element e holds K = k0 + 16*hb + e (contiguous, matches B)
    V16A a0, a1;
    const unsigned int* a32_0 = (const unsigned int*)(arow0 + k0);
    const unsigned int* a32_1 = (const unsigned int*)(arow1 + k0);
    const int ab = hb * 8;
#pragma unroll
    for (int j = 0; j < 4; ++j) {
      a0.u[j]     = a32_0[ab + j];
      a0.u[4 + j] = a32_0[4 + ab + j];
      a1.u[j]     = a32_1[ab + j];
      a1.u[4 + j] = a32_1[4 + ab + j];
    }
    // B 32x16 bf16 layout: element e holds K = k0 + 16*hb + e (contiguous 32B)
    v16bf b0 = *(const v16bf*)(brow0 + k0);
    v16bf b1 = *(const v16bf*)(brow1 + k0);
    acc[0][0] = __builtin_amdgcn_wmma_f32_16x16x32_bf16(false, a0.v, false, b0,
                                                        (short)0, acc[0][0], false, false);
    acc[0][1] = __builtin_amdgcn_wmma_f32_16x16x32_bf16(false, a0.v, false, b1,
                                                        (short)0, acc[0][1], false, false);
    acc[1][0] = __builtin_amdgcn_wmma_f32_16x16x32_bf16(false, a1.v, false, b0,
                                                        (short)0, acc[1][0], false, false);
    acc[1][1] = __builtin_amdgcn_wmma_f32_16x16x32_bf16(false, a1.v, false, b1,
                                                        (short)0, acc[1][1], false, false);
  }

#pragma unroll
  for (int ti = 0; ti < 2; ++ti) {
#pragma unroll
    for (int tj = 0; tj < 2; ++tj) {
      const int col = tn * 32 + tj * 16 + l16;
      const float bs = bias[col];
#pragma unroll
      for (int r = 0; r < 8; ++r) {
        const int row = tm * 32 + ti * 16 + r + 8 * hb;  // C: M = r + 8*(lane/16)
        const size_t idx = (size_t)row * N + col;
        float v = acc[ti][tj][r] + bs;
        if (MODE == 0) {
          outF[idx] = v;
          outB[idx] = f32_to_bf16(v);
        } else if (MODE == 1) {
          outB[idx] = f32_to_bf16(v);
        } else if (MODE == 2) {
          v += addF[idx];
          float g = 0.5f * v * (1.0f + erff(v * 0.70710678118654752f));  // exact GELU
          outB[idx] = f32_to_bf16(g);
        } else {  // MODE 5
          outF[idx] = addF[idx] + v;
        }
      }
    }
  }
}

// ---------------------------------------------------------------------------
// 5x5 neighborhood attention, 4 heads, dh=96.  One wave32 per (token, head).
// Lanes 0..24 own window positions (logits + softmax via shfl_xor); all 32
// lanes then accumulate output dims d = lane, lane+32, lane+64.
// qkv: (9216, 1152) bf16, column = c*384 + head*96 + d.
// out: (9216, 384) bf16.
// ---------------------------------------------------------------------------
#define HH 96
#define WW 96
#define NH 4
#define DH 96
#define KW 5

__global__ void natten_kernel(const unsigned short* __restrict__ qkv,
                              const float* __restrict__ rpb,
                              unsigned short* __restrict__ out) {
  const int lane = threadIdx.x & 31;
  const int wid  = (blockIdx.x * blockDim.x + threadIdx.x) >> 5;
  const int n = wid & (NH - 1);
  const int t = wid >> 2;
  const int h = t / WW, w = t - h * WW;
  int sh = h - 2; if (sh < 0) sh = 0; if (sh > HH - KW) sh = HH - KW;
  int sw = w - 2; if (sw < 0) sw = 0; if (sw > WW - KW) sw = WW - KW;

  const float scale = 0.1020620726159657f;  // 96^-0.5

  float logit = -3.4e38f;
  if (lane < KW * KW) {
    const int a = lane / KW, b = lane % KW;
    const int kh = sh + a, kw = sw + b;
    const int kt = kh * WW + kw;
    const unsigned short* qp = qkv + (size_t)t  * (3 * NH * DH) + n * DH;
    const unsigned short* kp = qkv + (size_t)kt * (3 * NH * DH) + (NH + n) * DH;
    float dot = 0.f;
#pragma unroll 4
    for (int d = 0; d < DH; ++d)
      dot += bf16_to_f32(qp[d]) * bf16_to_f32(kp[d]);
    const int bh = kh - h + (KW - 1);
    const int bw = kw - w + (KW - 1);
    logit = dot * scale + rpb[(n * (2 * KW - 1) + bh) * (2 * KW - 1) + bw];
  }

  // softmax across wave32
  float mx = logit;
  for (int off = 16; off > 0; off >>= 1) mx = fmaxf(mx, __shfl_xor(mx, off, 32));
  float e = (lane < KW * KW) ? __expf(logit - mx) : 0.f;
  float s = e;
  for (int off = 16; off > 0; off >>= 1) s += __shfl_xor(s, off, 32);
  const float attn = e / s;

  float acc0 = 0.f, acc1 = 0.f, acc2 = 0.f;
  for (int j = 0; j < KW * KW; ++j) {
    const float wgt = __shfl(attn, j, 32);
    const int a = j / KW, b = j % KW;
    const int kt = (sh + a) * WW + (sw + b);
    const unsigned short* vp = qkv + (size_t)kt * (3 * NH * DH) + (2 * NH + n) * DH;
    acc0 += wgt * bf16_to_f32(vp[lane]);
    acc1 += wgt * bf16_to_f32(vp[lane + 32]);
    acc2 += wgt * bf16_to_f32(vp[lane + 64]);
  }
  unsigned short* op = out + (size_t)t * (NH * DH) + n * DH;
  op[lane]      = f32_to_bf16(acc0);
  op[lane + 32] = f32_to_bf16(acc1);
  op[lane + 64] = f32_to_bf16(acc2);
}

// ---------------------------------------------------------------------------
// Host launcher
// ---------------------------------------------------------------------------
extern "C" void kernel_launch(void* const* d_in, const int* in_sizes, int n_in,
                              void* d_out, int out_size, void* d_ws, size_t ws_size,
                              hipStream_t stream) {
  (void)in_sizes; (void)n_in; (void)out_size; (void)ws_size;
  const float* x       = (const float*)d_in[0];
  const float* fc1_w   = (const float*)d_in[1];
  const float* fc1_b   = (const float*)d_in[2];
  const float* convA_w = (const float*)d_in[3];
  const float* convA_b = (const float*)d_in[4];
  const float* convB_w = (const float*)d_in[5];
  const float* convB_b = (const float*)d_in[6];
  const float* qkv_w   = (const float*)d_in[7];
  const float* qkv_b   = (const float*)d_in[8];
  const float* rpb     = (const float*)d_in[9];
  const float* proj_w  = (const float*)d_in[10];
  const float* proj_b  = (const float*)d_in[11];
  const float* fc2_w   = (const float*)d_in[12];
  const float* fc2_b   = (const float*)d_in[13];
  float* out = (float*)d_out;

  const int M = 96 * 96;  // 9216 tokens
  const int IN = 768, HID = 384, MID = 64, QKV = 3 * HID, OUT = 768;

  char* ws = (char*)d_ws;
  size_t off = 0;
  auto alloc = [&](size_t bytes) -> char* {
    char* p = ws + off;
    off += (bytes + 255) & ~(size_t)255;
    return p;
  };
  unsigned short* x_bf    = (unsigned short*)alloc((size_t)M * IN * 2);
  unsigned short* w1_bf   = (unsigned short*)alloc((size_t)HID * IN * 2);
  unsigned short* wA_bf   = (unsigned short*)alloc((size_t)MID * HID * 2);
  unsigned short* wB_bf   = (unsigned short*)alloc((size_t)HID * MID * 2);
  unsigned short* wq_bf   = (unsigned short*)alloc((size_t)QKV * HID * 2);
  unsigned short* wp_bf   = (unsigned short*)alloc((size_t)HID * HID * 2);
  unsigned short* w2_bf   = (unsigned short*)alloc((size_t)OUT * HID * 2);
  float*          xs_f32  = (float*)alloc((size_t)M * HID * 4);
  unsigned short* xs_bf   = (unsigned short*)alloc((size_t)M * HID * 2);
  unsigned short* a_bf    = (unsigned short*)alloc((size_t)M * MID * 2);
  unsigned short* g_bf    = (unsigned short*)alloc((size_t)M * HID * 2);
  unsigned short* qkv_bf  = (unsigned short*)alloc((size_t)M * QKV * 2);
  unsigned short* attn_bf = (unsigned short*)alloc((size_t)M * HID * 2);
  unsigned short* proj_bf = (unsigned short*)alloc((size_t)M * HID * 2);

  auto cast = [&](const float* s, unsigned short* d, int n) {
    int blocks = (n + 1023) / 1024;
    cast_f32_bf16<<<blocks, 256, 0, stream>>>(s, d, n);
  };
  cast(x, x_bf, M * IN);
  cast(fc1_w, w1_bf, HID * IN);
  cast(convA_w, wA_bf, MID * HID);
  cast(convB_w, wB_bf, HID * MID);
  cast(qkv_w, wq_bf, QKV * HID);
  cast(proj_w, wp_bf, HID * HID);
  cast(fc2_w, w2_bf, OUT * HID);

  auto nblk = [&](int N) { return ((M / 32) * (N / 32) + 7) / 8; };

  // fc1: xs = x @ fc1_w^T + b      (f32 + bf16 out)
  wmma_gemm<0><<<nblk(HID), 256, 0, stream>>>(x_bf, w1_bf, fc1_b, M, HID, IN,
                                              xs_f32, xs_bf, nullptr);
  // convA: a = xs @ convA_w^T + b  (bf16 out)
  wmma_gemm<1><<<nblk(MID), 256, 0, stream>>>(xs_bf, wA_bf, convA_b, M, MID, HID,
                                              nullptr, a_bf, nullptr);
  // convB + residual + GELU -> g   (bf16 out)
  wmma_gemm<2><<<nblk(HID), 256, 0, stream>>>(a_bf, wB_bf, convB_b, M, HID, MID,
                                              nullptr, g_bf, xs_f32);
  // qkv = g @ qkv_w^T + b          (bf16 out)
  wmma_gemm<1><<<nblk(QKV), 256, 0, stream>>>(g_bf, wq_bf, qkv_b, M, QKV, HID,
                                              nullptr, qkv_bf, nullptr);
  // neighborhood attention (wave32 per token/head): 36864 waves, 4 waves/block
  natten_kernel<<<(M * NH) / 4, 128, 0, stream>>>(qkv_bf, rpb, attn_bf);
  // proj
  wmma_gemm<1><<<nblk(HID), 256, 0, stream>>>(attn_bf, wp_bf, proj_b, M, HID, HID,
                                              nullptr, proj_bf, nullptr);
  // fc2 + final residual -> d_out (f32)
  wmma_gemm<5><<<nblk(OUT), 256, 0, stream>>>(proj_bf, w2_bf, fc2_b, M, OUT, HID,
                                              out, nullptr, x);
}
